// DecoderPreLN2_59090160059091
// MI455X (gfx1250) — compile-verified, matrix operand order checked by reading, not run on hardware
//
#include <hip/hip_runtime.h>
#include <hip/hip_bf16.h>

typedef __attribute__((ext_vector_type(16))) __bf16 v16bf;
typedef __attribute__((ext_vector_type(8)))  __bf16 v8bf;
typedef __attribute__((ext_vector_type(4)))  __bf16 v4bf;
typedef __attribute__((ext_vector_type(8)))  float  v8f;
typedef __attribute__((ext_vector_type(4)))  unsigned int u32x4;
typedef __attribute__((ext_vector_type(8)))  unsigned int u32x8;

#define NB 8
#define TQ 512
#define TK 1024
#define DM 1024
#define NH 16
#define HD 64
#define NL 4

__device__ __forceinline__ v8f wmma_bf16(v16bf a, v16bf b, v8f c) {
  // D = A*B + C, f32 accumulate, 16x16x32 bf16
  return __builtin_amdgcn_wmma_f32_16x16x32_bf16(false, a, false, b, (short)0, c,
                                                 false, false);
}

// Load a 16x32 bf16 A/B fragment from a row-major (row = M or N, contiguous K)
// source. Per ISA layout: lanes 0-15 hold K 0-7 (VGPR0-3) and 16-23 (VGPR4-7);
// lanes 16-31 hold K 8-15 and 24-31. Two 16-byte contiguous loads per lane.
__device__ __forceinline__ v16bf load_frag(const __bf16* base, int ld, int row0,
                                           int k0, int lane) {
  const int r  = row0 + (lane & 15);
  const int kc = k0 + ((lane & 16) ? 8 : 0);
  const __bf16* p = base + (long)r * ld + kc;
  v8bf lo = *(const v8bf*)p;
  v8bf hi = *(const v8bf*)(p + 16);
  v16bf f;
#pragma unroll
  for (int i = 0; i < 8; ++i) { f[i] = lo[i]; f[i + 8] = hi[i]; }
  return f;
}

// Issue a TDM 2D tile DMA: global (row-major, bf16, row stride `stride` elems)
// -> LDS at byte offset lds_off, tile = tile_w x tile_h elements, packed
// contiguously in LDS. Per-wave scalar op, tracked with TENSORcnt.
// D# group0: count=1 | lds_addr | global_addr[56:0] | type=2.
// D# group1: data_size=1(2B); tensor_dim0/1 = 1<<20 (no OOB clip);
//            tile_dim0/1; tensor_dim0_stride.
__device__ __forceinline__ void tdm_load_tile_2d(unsigned lds_off, const void* g,
                                                 unsigned tile_w, unsigned tile_h,
                                                 unsigned stride) {
  const unsigned long long ga = (unsigned long long)(uintptr_t)g;
  u32x4 g0;
  g0[0] = 1u;                                                 // count=1 (valid)
  g0[1] = lds_off;                                            // LDS byte addr
  g0[2] = (unsigned)ga;                                       // global addr lo
  g0[3] = ((unsigned)(ga >> 32) & 0x01ffffffu) | (2u << 30);  // addr hi | type=2
  u32x8 g1;
  g1[0] = 0x00010000u;              // workgroup_mask=0, data_size=1 (2 bytes)
  g1[1] = 0u;                       // atomic_barrier=0, tensor_dim0 lo16 = 0
  g1[2] = 0x10u;                    // tensor_dim0 = 1<<20 ; tensor_dim1 lo16=0
  g1[3] = (tile_w << 16) | 0x10u;   // tile_dim0 ; tensor_dim1 = 1<<20
  g1[4] = tile_h;                   // tile_dim1 (tile_dim2 = 0)
  g1[5] = stride;                   // tensor_dim0_stride[31:0]
  g1[6] = 0u;                       // stride hi, dim1_stride lo
  g1[7] = 0u;
  asm volatile("tensor_load_to_lds %0, %1" :: "s"(g0), "s"(g1) : "memory");
}

// ---------------- one-time conversion kernels ----------------

__global__ void conv_enc(const float* __restrict__ E, __bf16* __restrict__ Eb) {
  const long i = ((long)blockIdx.x * 256 + threadIdx.x) * 4;
  const float4 v = *(const float4*)(E + i);
  v4bf o;
  o[0] = (__bf16)v.x; o[1] = (__bf16)v.y; o[2] = (__bf16)v.z; o[3] = (__bf16)v.w;
  *(v4bf*)(Eb + i) = o;
}

// Wt[n*D + k] = (bf16) W[k*D + n]   (per layer in blockIdx.z)
__global__ void conv_w(const float* __restrict__ W, __bf16* __restrict__ Wt) {
  __shared__ float tile[32][33];
  const float* w  = W  + (long)blockIdx.z * DM * DM;
  __bf16*      wt = Wt + (long)blockIdx.z * DM * DM;
  const int k0 = blockIdx.x * 32, n0 = blockIdx.y * 32;
  const int tx = threadIdx.x, ty = threadIdx.y;
#pragma unroll
  for (int i = 0; i < 32; i += 8)
    tile[ty + i][tx] = w[(long)(k0 + ty + i) * DM + n0 + tx];
  __syncthreads();
#pragma unroll
  for (int i = 0; i < 32; i += 8)
    wt[(long)(n0 + ty + i) * DM + k0 + tx] = (__bf16)tile[tx][ty + i];
}

__global__ void gather_emb(const int* __restrict__ ids,
                           const float* __restrict__ emb, float* __restrict__ X) {
  const long row = blockIdx.x;
  const long id  = ids[row];
  const int  c   = threadIdx.x * 4;
  *(float4*)(X + row * DM + c) = *(const float4*)(emb + id * DM + c);
}

// ---------------- per-layer kernels ----------------

__global__ void __launch_bounds__(256)
ln_bf16(const float* __restrict__ X, const float* __restrict__ g,
        const float* __restrict__ bta, __bf16* __restrict__ Hh) {
  __shared__ float red[256];
  const int  t   = threadIdx.x;
  const long row = blockIdx.x;
  const float* x = X + row * DM;
  float v[4];
  float s = 0.f;
#pragma unroll
  for (int i = 0; i < 4; ++i) { v[i] = x[t + 256 * i]; s += v[i]; }
  red[t] = s; __syncthreads();
  for (int off = 128; off > 0; off >>= 1) {
    if (t < off) red[t] += red[t + off];
    __syncthreads();
  }
  const float mu = red[0] * (1.f / 1024.f);
  __syncthreads();
  float ss = 0.f;
#pragma unroll
  for (int i = 0; i < 4; ++i) { const float d = v[i] - mu; ss += d * d; }
  red[t] = ss; __syncthreads();
  for (int off = 128; off > 0; off >>= 1) {
    if (t < off) red[t] += red[t + off];
    __syncthreads();
  }
  const float inv = rsqrtf(red[0] * (1.f / 1024.f) + 1e-5f);
#pragma unroll
  for (int i = 0; i < 4; ++i) {
    const int c = t + 256 * i;
    Hh[row * DM + c] = (__bf16)((v[i] - mu) * inv * g[c] + bta[c]);
  }
}

// C[M,1024] = A[M,1024] @ W + bias.  Bt is the weight pre-transposed to [N,K].
// Block tile 128x128, 8 waves (4x2), wave tile 32x64, K-step 32.
// Tiles are DMA'd into double-buffered LDS by the Tensor Data Mover
// (tensor_load_to_lds, TENSORcnt), overlapping DMA of tile k+1 with WMMA on k.
// mode 0: C row-major bf16 [m*1024+n].
// mode 1: V head-transpose store: Vt[((b*16+h)*64+d)*1024 + t]
__global__ void __launch_bounds__(256)
proj_gemm(const __bf16* __restrict__ A, const __bf16* __restrict__ Bt,
          const float* __restrict__ bias, __bf16* __restrict__ C, int mode) {
  __shared__ __bf16 As[2][128 * 32];
  __shared__ __bf16 Bs[2][128 * 32];
  const int t = threadIdx.x;
  const int lane = t & 31, w = t >> 5;
  const int wm = w & 3, wn = w >> 2;
  const int m0 = blockIdx.x * 128, n0 = blockIdx.y * 128;

  const __bf16* gA = A  + (long)m0 * DM;   // tile origin; k advances by 32
  const __bf16* gB = Bt + (long)n0 * DM;

  const v8f vzero = {0.f, 0.f, 0.f, 0.f, 0.f, 0.f, 0.f, 0.f};
  v8f acc[2][4];
#pragma unroll
  for (int i = 0; i < 2; ++i)
#pragma unroll
    for (int j = 0; j < 4; ++j) acc[i][j] = vzero;

  // Prologue: wave 0 kicks off the first pair of tile DMAs.
  if (w == 0) {
    tdm_load_tile_2d((unsigned)(uintptr_t)&As[0][0], gA, 32, 128, DM);
    tdm_load_tile_2d((unsigned)(uintptr_t)&Bs[0][0], gB, 32, 128, DM);
  }

  const int NK = DM / 32;
  for (int kk = 0; kk < NK; ++kk) {
    const int cur = kk & 1, nxt = cur ^ 1;
    if (w == 0) {
      if (kk + 1 < NK) {
        tdm_load_tile_2d((unsigned)(uintptr_t)&As[nxt][0], gA + (kk + 1) * 32,
                         32, 128, DM);
        tdm_load_tile_2d((unsigned)(uintptr_t)&Bs[nxt][0], gB + (kk + 1) * 32,
                         32, 128, DM);
        // 4 outstanding; tensor ops complete in order -> <=2 means current
        // buffer's two DMAs are done while next buffer's stay in flight.
        asm volatile("s_wait_tensorcnt 0x2" ::: "memory");
      } else {
        asm volatile("s_wait_tensorcnt 0x0" ::: "memory");
      }
    }
    __syncthreads();   // publish LDS tile `cur` to all waves

    v16bf af[2], bfr[4];
#pragma unroll
    for (int i = 0; i < 2; ++i)
      af[i] = load_frag(&As[cur][0], 32, wm * 32 + 16 * i, 0, lane);
#pragma unroll
    for (int j = 0; j < 4; ++j)
      bfr[j] = load_frag(&Bs[cur][0], 32, wn * 64 + 16 * j, 0, lane);
#pragma unroll
    for (int i = 0; i < 2; ++i)
#pragma unroll
      for (int j = 0; j < 4; ++j) acc[i][j] = wmma_bf16(af[i], bfr[j], acc[i][j]);
    __syncthreads();   // all waves done with `cur` before it is re-DMAed
  }

  const int cl = lane & 15;
  const int rh = (lane >> 4) << 3;
#pragma unroll
  for (int i = 0; i < 2; ++i)
#pragma unroll
    for (int j = 0; j < 4; ++j) {
      const int gn = n0 + wn * 64 + 16 * j + cl;
      const float bv = bias[gn];
#pragma unroll
      for (int e = 0; e < 8; ++e) {
        const int gm = m0 + wm * 32 + 16 * i + rh + e;
        const float val = acc[i][j][e] + bv;
        if (mode == 0) {
          C[(long)gm * DM + gn] = (__bf16)val;
        } else {
          const int b = gm >> 10, tk = gm & (TK - 1);
          const int hh = gn >> 6, d = gn & (HD - 1);
          C[(((long)(b * NH + hh) * HD + d) << 10) + tk] = (__bf16)val;
        }
      }
    }
}

// Fused attention for one (batch, head): 16 query rows per block.
// Phase 1: logits = Q@K^T*scale+mask (wave w owns 128 keys).
// Phase 2: row softmax via shfl_xor + cross-wave LDS reduction.
// Phase 3: P (bf16) into LDS; PV via WMMA with split-K across waves.
__global__ void __launch_bounds__(256)
fused_attn(const __bf16* __restrict__ Q, const __bf16* __restrict__ Kc,
           const __bf16* __restrict__ Vt, const float* __restrict__ mask,
           float* __restrict__ X) {
  __shared__ __bf16 Pb[16 * TK];
  __shared__ float  redw[8][16];
  __shared__ float  grow[16];
  __shared__ float  outred[2][16][HD];

  const int t = threadIdx.x;
  const int lane = t & 31, w = t >> 5;
  const int bh = blockIdx.y, b = bh >> 4, h = bh & (NH - 1);
  const int m0 = blockIdx.x * 16;
  const int cl = lane & 15;
  const int rh = (lane >> 4) << 3;

  const __bf16* Aq = Q  + ((long)b * TQ + m0) * DM + h * HD;
  const __bf16* Bk = Kc + ((long)b * TK) * DM + h * HD;

  const v8f vzero = {0.f, 0.f, 0.f, 0.f, 0.f, 0.f, 0.f, 0.f};
  v8f acc[8];
#pragma unroll
  for (int j = 0; j < 8; ++j) acc[j] = vzero;

#pragma unroll
  for (int k0 = 0; k0 < HD; k0 += 32) {
    v16bf a = load_frag(Aq, DM, 0, k0, lane);
#pragma unroll
    for (int j = 0; j < 8; ++j) {
      v16bf bf = load_frag(Bk, DM, w * 128 + 16 * j, k0, lane);
      acc[j] = wmma_bf16(a, bf, acc[j]);
    }
  }

  const float scale = 0.125f;  // 1/sqrt(64)
#pragma unroll
  for (int j = 0; j < 8; ++j)
#pragma unroll
    for (int e = 0; e < 8; ++e)
      acc[j][e] = acc[j][e] * scale + mask[b * TK + w * 128 + 16 * j + cl];

  // ---- row max (each e is a distinct row; reduce over 16 lanes of the half)
  float rmax[8];
#pragma unroll
  for (int e = 0; e < 8; ++e) {
    float m = -3.0e38f;
#pragma unroll
    for (int j = 0; j < 8; ++j) m = fmaxf(m, acc[j][e]);
#pragma unroll
    for (int s = 1; s < 16; s <<= 1) m = fmaxf(m, __shfl_xor(m, s, 32));
    rmax[e] = m;
  }
  if (cl == 0) {
#pragma unroll
    for (int e = 0; e < 8; ++e) redw[w][rh + e] = rmax[e];
  }
  __syncthreads();
  if (t < 16) {
    float m = -3.0e38f;
    for (int ww = 0; ww < 8; ++ww) m = fmaxf(m, redw[ww][t]);
    grow[t] = m;
  }
  __syncthreads();
  float gm8[8];
#pragma unroll
  for (int e = 0; e < 8; ++e) gm8[e] = grow[rh + e];
  __syncthreads();

  // ---- exp + row sum
  float rsum[8];
#pragma unroll
  for (int e = 0; e < 8; ++e) {
    float s = 0.f;
#pragma unroll
    for (int j = 0; j < 8; ++j) {
      acc[j][e] = __expf(acc[j][e] - gm8[e]);
      s += acc[j][e];
    }
#pragma unroll
    for (int sh = 1; sh < 16; sh <<= 1) s += __shfl_xor(s, sh, 32);
    rsum[e] = s;
  }
  if (cl == 0) {
#pragma unroll
    for (int e = 0; e < 8; ++e) redw[w][rh + e] = rsum[e];
  }
  __syncthreads();
  if (t < 16) {
    float s = 0.f;
    for (int ww = 0; ww < 8; ++ww) s += redw[ww][t];
    grow[t] = s;
  }
  __syncthreads();

  // ---- normalized probabilities into LDS (bf16)
#pragma unroll
  for (int e = 0; e < 8; ++e) {
    const float inv = 1.f / grow[rh + e];
#pragma unroll
    for (int j = 0; j < 8; ++j)
      Pb[(rh + e) * TK + w * 128 + 16 * j + cl] = (__bf16)(acc[j][e] * inv);
  }
  __syncthreads();

  // ---- PV: out[16,64]; wave w -> n-subtile (w&3), K half (w>>2)
  const int j2 = w & 3, kh = w >> 2;
  const __bf16* Bv = Vt + (long)bh * HD * TK;  // rows = d, contiguous k
  v8f oacc = vzero;
  for (int k0 = kh * 512; k0 < kh * 512 + 512; k0 += 32) {
    v16bf a  = load_frag(Pb, TK, 0, k0, lane);
    v16bf bf = load_frag(Bv, TK, 16 * j2, k0, lane);
    oacc = wmma_bf16(a, bf, oacc);
  }
#pragma unroll
  for (int e = 0; e < 8; ++e) outred[kh][rh + e][16 * j2 + cl] = oacc[e];
  __syncthreads();
  for (int idx = t; idx < 16 * HD; idx += 256) {
    const int r = idx >> 6, c = idx & (HD - 1);
    X[((long)b * TQ + m0 + r) * DM + h * HD + c] =
        outred[0][r][c] + outred[1][r][c];
  }
}

// ---------------- host launcher ----------------

extern "C" void kernel_launch(void* const* d_in, const int* in_sizes, int n_in,
                              void* d_out, int out_size, void* d_ws, size_t ws_size,
                              hipStream_t stream) {
  const int*   ids  = (const int*)d_in[0];
  const float* enc  = (const float*)d_in[1];
  const float* mask = (const float*)d_in[2];
  const float* emb  = (const float*)d_in[3];
  const float* ln_g = (const float*)d_in[4];
  const float* ln_b = (const float*)d_in[5];
  const float* wq   = (const float*)d_in[6];
  const float* bq   = (const float*)d_in[7];
  const float* wk   = (const float*)d_in[8];
  const float* bk   = (const float*)d_in[9];
  const float* wv   = (const float*)d_in[10];
  const float* bv   = (const float*)d_in[11];
  float* out = (float*)d_out;

  char* p = (char*)d_ws;
  auto carve = [&](size_t bytes) -> char* {
    char* r = p;
    p += (bytes + 255) & ~(size_t)255;
    return r;
  };
  __bf16* enc_b = (__bf16*)carve((size_t)NB * TK * DM * 2);  // bf16 encoder
  __bf16* wqt   = (__bf16*)carve((size_t)NL * DM * DM * 2);  // W^T bf16
  __bf16* wkt   = (__bf16*)carve((size_t)NL * DM * DM * 2);
  __bf16* wvt   = (__bf16*)carve((size_t)NL * DM * DM * 2);
  float*  x     = (float*) carve((size_t)NB * TQ * DM * 4);  // activations f32
  __bf16* hb    = (__bf16*)carve((size_t)NB * TQ * DM * 2);  // LN output
  __bf16* qb    = (__bf16*)carve((size_t)NB * TQ * DM * 2);  // Q
  __bf16* kb    = (__bf16*)carve((size_t)NB * TK * DM * 2);  // K
  __bf16* vt    = (__bf16*)carve((size_t)NB * TK * DM * 2);  // V head-transposed

  conv_enc<<<(NB * TK * DM) / (256 * 4), 256, 0, stream>>>(enc, enc_b);
  conv_w<<<dim3(32, 32, NL), dim3(32, 8), 0, stream>>>(wq, wqt);
  conv_w<<<dim3(32, 32, NL), dim3(32, 8), 0, stream>>>(wk, wkt);
  conv_w<<<dim3(32, 32, NL), dim3(32, 8), 0, stream>>>(wv, wvt);
  gather_emb<<<NB * TQ, 256, 0, stream>>>(ids, emb, x);

  for (int l = 0; l < NL; ++l) {
    ln_bf16<<<NB * TQ, 256, 0, stream>>>(x, ln_g + l * DM, ln_b + l * DM, hb);
    proj_gemm<<<dim3((NB * TQ) / 128, DM / 128), 256, 0, stream>>>(
        hb, wqt + (size_t)l * DM * DM, bq + l * DM, qb, 0);
    proj_gemm<<<dim3((NB * TK) / 128, DM / 128), 256, 0, stream>>>(
        enc_b, wkt + (size_t)l * DM * DM, bk + l * DM, kb, 0);
    proj_gemm<<<dim3((NB * TK) / 128, DM / 128), 256, 0, stream>>>(
        enc_b, wvt + (size_t)l * DM * DM, bv + l * DM, vt, 1);
    fused_attn<<<dim3(TQ / 16, NB * NH), 256, 0, stream>>>(
        qb, kb, vt, mask, (l == NL - 1) ? out : x);
  }
  (void)in_sizes; (void)n_in; (void)out_size; (void)ws_size;
}
